// SelfAttention_86818468922167
// MI455X (gfx1250) — matrix operand order repaired and checked by reading, not compiled, not run on hardware
//
#include <hip/hip_runtime.h>
#include <hip/hip_bf16.h>

typedef __attribute__((ext_vector_type(16))) __bf16 v16bf;
typedef __attribute__((ext_vector_type(8)))  __bf16 v8bf;
typedef __attribute__((ext_vector_type(8)))  float  v8f;
typedef __attribute__((ext_vector_type(4)))  unsigned int u32x4;
typedef __attribute__((ext_vector_type(8)))  int          i32x8;
typedef __attribute__((ext_vector_type(4)))  int          i32x4;

// Problem constants: B=8, S=2048, D=512, M = B*S = 16384
#define ATT_S 2048
#define ATT_D 512
#define ATT_M 16384

#if defined(__has_builtin)
#if __has_builtin(__builtin_amdgcn_tensor_load_to_lds) && \
    __has_builtin(__builtin_amdgcn_s_wait_tensorcnt)
#define HAVE_TDM 1
#endif
#endif
#ifndef HAVE_TDM
#define HAVE_TDM 0
#endif

static __device__ __forceinline__ v8f wmma_bf16(v16bf a, v16bf b, v8f c) {
  // v_wmma_f32_16x16x32_bf16 : D(16x16 f32) = A(16x32 bf16) x B(32x16 bf16) + C
  return __builtin_amdgcn_wmma_f32_16x16x32_bf16(false, a, false, b, (short)0, c,
                                                 false, false);
}

// A-fragment (16x32 bf16, row-major source with row stride lda elements).
// Lane l: M = l&15 ; lane-half 0 holds K {0..15}, half 1 holds K {16..31}.
static __device__ __forceinline__ v16bf make_a_frag(const __bf16* rowbase, int lda, int lane) {
  const int m  = lane & 15;
  const int kb = (lane >> 4) << 4;            // 0 or 16
  const __bf16* p = rowbase + m * lda;
  v8bf lo = *(const v8bf*)(p + kb);           // K = kb .. kb+7
  v8bf hi = *(const v8bf*)(p + kb + 8);       // K = kb+8 .. kb+15
  return __builtin_shufflevector(lo, hi, 0,1,2,3,4,5,6,7,8,9,10,11,12,13,14,15);
}

// ---------------------------------------------------------------------------
// Kernel 1: Y = x @ W + bias, output bf16.
//   transposed==0 : out[m][e]            (row-major [M][D])   -> Q, K
//   transposed==1 : out[(b*D+e)][s]      (per-batch transpose) -> Vt
// Tile: 128 (M) x 64 (N) per workgroup, K-steps of 32. 8 waves, 4 WMMA tiles each.
// ---------------------------------------------------------------------------
__global__ __launch_bounds__(256)
void qkv_proj_kernel(const float* __restrict__ x, const float* __restrict__ W,
                     const float* __restrict__ bias, unsigned short* __restrict__ out_raw,
                     int transposed) {
  __bf16* __restrict__ out = reinterpret_cast<__bf16*>(out_raw);

  __shared__ __attribute__((aligned(32))) __bf16 sA[128][32];   // x tile (bf16)
  __shared__ __attribute__((aligned(32))) __bf16 sB[64][32];    // W tile, transposed [e][d]

  const int tid  = threadIdx.x;
  const int lane = tid & 31;
  const int wv   = tid >> 5;
  const int mblk = blockIdx.x * 128;
  const int nblk = blockIdx.y * 64;

  const int wm = wv >> 1;                 // 0..3 : 32-row strip
  const int wn = wv & 1;                  // 0..1 : 32-col strip

  v8f zero{};
  v8f acc00 = zero, acc01 = zero, acc10 = zero, acc11 = zero;

  const int arow = tid >> 1;              // 0..127
  const int acol = (tid & 1) * 16;        // 0 or 16
  const int brow = tid >> 3;              // 0..31  (d within K-step)
  const int bcol = (tid & 7) * 8;         // 0..56  (e within tile)

  for (int k0 = 0; k0 < ATT_D; k0 += 32) {
    // Stage A: x[mblk+arow][k0 + acol .. +15] -> bf16 LDS (packed 16B stores)
#pragma unroll
    for (int i = 0; i < 2; ++i) {
      const float4 v0 = *(const float4*)(x + (size_t)(mblk + arow) * ATT_D + k0 + acol + i * 8);
      const float4 v1 = *(const float4*)(x + (size_t)(mblk + arow) * ATT_D + k0 + acol + i * 8 + 4);
      v8bf pk;
      pk[0] = (__bf16)v0.x; pk[1] = (__bf16)v0.y; pk[2] = (__bf16)v0.z; pk[3] = (__bf16)v0.w;
      pk[4] = (__bf16)v1.x; pk[5] = (__bf16)v1.y; pk[6] = (__bf16)v1.z; pk[7] = (__bf16)v1.w;
      *(v8bf*)(&sA[arow][acol + i * 8]) = pk;
    }
    // Stage B transposed: W[k0+brow][nblk+bcol+j] -> sB[bcol+j][brow]
#pragma unroll
    for (int i = 0; i < 2; ++i) {
      const float4 v = *(const float4*)(W + (size_t)(k0 + brow) * ATT_D + nblk + bcol + i * 4);
      sB[bcol + i * 4 + 0][brow] = (__bf16)v.x;
      sB[bcol + i * 4 + 1][brow] = (__bf16)v.y;
      sB[bcol + i * 4 + 2][brow] = (__bf16)v.z;
      sB[bcol + i * 4 + 3][brow] = (__bf16)v.w;
    }
    __syncthreads();

    const v16bf a0 = make_a_frag(&sA[wm * 32][0], 32, lane);
    const v16bf a1 = make_a_frag(&sA[wm * 32 + 16][0], 32, lane);
    const int bn = lane & 15;
    const int kh = (lane >> 4) << 4;      // 0 or 16
    const v16bf b0 = *(const v16bf*)(&sB[wn * 32 + bn][kh]);
    const v16bf b1 = *(const v16bf*)(&sB[wn * 32 + 16 + bn][kh]);

    acc00 = wmma_bf16(a0, b0, acc00);
    acc01 = wmma_bf16(a0, b1, acc01);
    acc10 = wmma_bf16(a1, b0, acc10);
    acc11 = wmma_bf16(a1, b1, acc11);
    __syncthreads();
  }

  // Epilogue: + bias, convert to bf16, store
  const int cn = lane & 15;
  const int rh = (lane >> 4) << 3;        // 0 or 8
  const int e0 = nblk + wn * 32 + cn;
  const int e1 = e0 + 16;
  const float bias0 = bias[e0];
  const float bias1 = bias[e1];
#pragma unroll
  for (int r = 0; r < 8; ++r) {
    const int m0 = mblk + wm * 32 + rh + r;
    const int m1 = m0 + 16;
    const float v00 = acc00[r] + bias0;
    const float v01 = acc01[r] + bias1;
    const float v10 = acc10[r] + bias0;
    const float v11 = acc11[r] + bias1;
    if (!transposed) {
      out[(size_t)m0 * ATT_D + e0] = (__bf16)v00;
      out[(size_t)m0 * ATT_D + e1] = (__bf16)v01;
      out[(size_t)m1 * ATT_D + e0] = (__bf16)v10;
      out[(size_t)m1 * ATT_D + e1] = (__bf16)v11;
    } else {
      const int b0i = m0 >> 11, s0 = m0 & 2047;
      const int b1i = m1 >> 11, s1 = m1 & 2047;
      out[((size_t)b0i * ATT_D + e0) * ATT_S + s0] = (__bf16)v00;
      out[((size_t)b0i * ATT_D + e1) * ATT_S + s0] = (__bf16)v01;
      out[((size_t)b1i * ATT_D + e0) * ATT_S + s1] = (__bf16)v10;
      out[((size_t)b1i * ATT_D + e1) * ATT_S + s1] = (__bf16)v11;
    }
  }
}

// ---------------------------------------------------------------------------
// Kernel 2: flash attention (no 1/sqrt(d) scaling, matching reference).
// Workgroup: 32 query rows, 8 waves, key blocks of 64.
//   Q block staged to LDS via Tensor Data Mover (2D tile descriptor) when
//   available, else vector-copy fallback.
//   Phase A: wave (qg = w&1, kt = w>>1) computes 16x16 score tile -> sS (f32)
//   Phase B: cooperative online softmax over sS -> sP (bf16), alpha, l
//   Phase C: wave (qg = w&1, dblk = w>>1): accO = alpha*accO + P(16x64)@V(64x128)
// ---------------------------------------------------------------------------
__global__ __launch_bounds__(256)
void flash_attn_kernel(const unsigned short* __restrict__ Qr,
                       const unsigned short* __restrict__ Kr,
                       const unsigned short* __restrict__ Vtr,
                       float* __restrict__ out) {
  const __bf16* __restrict__ Q  = reinterpret_cast<const __bf16*>(Qr);
  const __bf16* __restrict__ K  = reinterpret_cast<const __bf16*>(Kr);
  const __bf16* __restrict__ Vt = reinterpret_cast<const __bf16*>(Vtr);

  __shared__ __attribute__((aligned(32))) __bf16 sQ[32][ATT_D];  // 32 KB
  __shared__ __attribute__((aligned(32))) float  sS[32][64];     // 8 KB
  __shared__ __attribute__((aligned(32))) __bf16 sP[32][64];     // 4 KB
  __shared__ float s_m[32], s_l[32], s_al[32];
  __shared__ float s_red[32][8];

  const int tid  = threadIdx.x;
  const int lane = tid & 31;
  const int wv   = tid >> 5;
  const int bb   = blockIdx.y;
  const int qblk = blockIdx.x * 32;

  // ---- Stage Q block (32 x 512 bf16 contiguous tile) into LDS ----
  const size_t gq0 = ((size_t)bb * ATT_S + qblk) * ATT_D;
#if HAVE_TDM
  if (wv == 0) {
    // Tensor DMA descriptor: 2D tile 512 x 32 (elem = 2B) from row-major Q.
    const unsigned lds_addr   = (unsigned)(uintptr_t)(&sQ[0][0]);
    const unsigned long long ga = (unsigned long long)(uintptr_t)(Q + gq0);
    u32x4 g0;
    g0[0] = 1u;                                          // count=1 (user D#)
    g0[1] = lds_addr;                                    // LDS byte address
    g0[2] = (unsigned)(ga & 0xFFFFFFFFu);                // global_addr[31:0]
    g0[3] = (unsigned)((ga >> 32) & 0x1FFFFFFu) | (2u << 30); // addr[56:32] | type=2
    i32x8 g1;
    g1[0] = (int)(1u << 16);                             // data_size = 1 (2 bytes)
    g1[1] = (int)(512u << 16);                           // tensor_dim0[15:0] = 512
    g1[2] = (int)((16384u & 0xFFFFu) << 16);             // dim0 hi=0 | tensor_dim1 lo = 16384
    g1[3] = (int)(512u << 16);                           // tensor_dim1 hi=0 | tile_dim0 = 512
    g1[4] = 32;                                          // tile_dim1 = 32, tile_dim2 = 0
    g1[5] = 512;                                         // tensor_dim0_stride[31:0] = 512
    g1[6] = 0;                                           // stride hi | dim1_stride lo
    g1[7] = 0;
    i32x4 gz4 = {0, 0, 0, 0};
    i32x8 gz8 = {0, 0, 0, 0, 0, 0, 0, 0};
    __builtin_amdgcn_tensor_load_to_lds(g0, g1, gz4, gz4, gz8, 0);
    __builtin_amdgcn_s_wait_tensorcnt(0);
  }
#else
  {
    const int row = tid >> 3;
    const int seg = tid & 7;
    const size_t gq = gq0 + (size_t)row * ATT_D;
#pragma unroll
    for (int i = 0; i < 8; ++i) {
      const int c = (seg * 8 + i) * 8;                   // 8 bf16 = 16B chunks
      *(uint4*)(&sQ[row][c]) = *(const uint4*)(Q + gq + c);
    }
  }
#endif
  if (tid < 32) { s_m[tid] = -3.0e38f; s_l[tid] = 0.0f; }
  __syncthreads();

  const int qg   = wv & 1;             // query group (rows qg*16 .. +15)
  const int kt   = wv >> 1;            // key sub-tile in phase A
  const int dblk = (wv >> 1) * 128;    // d columns in phase C
  const int srow = tid >> 3;           // softmax: row
  const int sseg = tid & 7;            // softmax: 8-col segment

  const int n  = lane & 15;
  const int kh = (lane >> 4) << 4;     // 0 or 16 (B-frag K half)
  const int rh = (lane >> 4) << 3;     // 0 or 8  (C/D row half)

  v8f zero{};
  v8f accO[8];
#pragma unroll
  for (int i = 0; i < 8; ++i) accO[i] = zero;

  for (int kb = 0; kb < ATT_S; kb += 64) {
    // ---- Phase A: 16x16 score tile per wave ----
    v8f sc = zero;
    const size_t krow = ((size_t)bb * ATT_S + kb + kt * 16 + n) * ATT_D;
    if (kb + 64 < ATT_S)   // pull next key block's row toward L2/L0
      __builtin_prefetch(K + krow + (size_t)64 * ATT_D + kh, 0, 3);
#pragma unroll 4
    for (int dc = 0; dc < ATT_D; dc += 32) {
      const v16bf a   = make_a_frag(&sQ[qg * 16][dc], ATT_D, lane);
      const v16bf bkf = *(const v16bf*)(K + krow + dc + kh);   // contiguous 32B / lane
      sc = wmma_bf16(a, bkf, sc);
    }
    {
      const int rr = qg * 16 + rh;
      const int cc = kt * 16 + n;
#pragma unroll
      for (int r = 0; r < 8; ++r) sS[rr + r][cc] = sc[r];
    }
    __syncthreads();

    // ---- Phase B: online softmax ----
    float vals[8];
    float mx = -3.0e38f;
#pragma unroll
    for (int j = 0; j < 8; ++j) { vals[j] = sS[srow][sseg * 8 + j]; mx = fmaxf(mx, vals[j]); }
    s_red[srow][sseg] = mx;
    __syncthreads();
    if (tid < 32) {
      float tm = s_red[tid][0];
#pragma unroll
      for (int s = 1; s < 8; ++s) tm = fmaxf(tm, s_red[tid][s]);
      const float mo = s_m[tid];
      const float mnw = fmaxf(mo, tm);
      s_al[tid] = __expf(mo - mnw);
      s_m[tid]  = mnw;
    }
    __syncthreads();
    const float mrow = s_m[srow];
    float psum = 0.0f;
#pragma unroll
    for (int j = 0; j < 8; ++j) {
      const float p = __expf(vals[j] - mrow);
      sP[srow][sseg * 8 + j] = (__bf16)p;
      psum += p;
    }
    s_red[srow][sseg] = psum;
    __syncthreads();
    if (tid < 32) {
      float t = 0.0f;
#pragma unroll
      for (int s = 0; s < 8; ++s) t += s_red[tid][s];
      s_l[tid] = s_l[tid] * s_al[tid] + t;
    }
    __syncthreads();

    // ---- Phase C: accO = alpha * accO + P @ V ----
    float al[8];
#pragma unroll
    for (int r = 0; r < 8; ++r) al[r] = s_al[qg * 16 + rh + r];
#pragma unroll
    for (int nt = 0; nt < 8; ++nt)
#pragma unroll
      for (int r = 0; r < 8; ++r) accO[nt][r] *= al[r];

#pragma unroll
    for (int kc = 0; kc < 2; ++kc) {
      const v16bf ap = make_a_frag(&sP[qg * 16][kc * 32], 64, lane);
#pragma unroll
      for (int nt = 0; nt < 8; ++nt) {
        const size_t vrow = ((size_t)bb * ATT_D + dblk + nt * 16 + n) * ATT_S;
        if (kc == 0 && kb + 64 < ATT_S)
          __builtin_prefetch(Vt + vrow + kb + 64, 0, 3);
        const v16bf bv = *(const v16bf*)(Vt + vrow + kb + kc * 32 + kh);
        accO[nt] = wmma_bf16(ap, bv, accO[nt]);
      }
    }
    __syncthreads();   // protect sP/s_al from next iteration's phase B
  }

  // ---- Final: normalize by 1/l, store fp32 ----
  float linv[8];
#pragma unroll
  for (int r = 0; r < 8; ++r) linv[r] = 1.0f / s_l[qg * 16 + rh + r];
#pragma unroll
  for (int nt = 0; nt < 8; ++nt) {
    const int e = dblk + nt * 16 + n;
#pragma unroll
    for (int r = 0; r < 8; ++r) {
      const size_t row = (size_t)bb * ATT_S + qblk + qg * 16 + rh + r;
      out[row * ATT_D + e] = accO[nt][r] * linv[r];
    }
  }
}

// ---------------------------------------------------------------------------
extern "C" void kernel_launch(void* const* d_in, const int* in_sizes, int n_in,
                              void* d_out, int out_size, void* d_ws, size_t ws_size,
                              hipStream_t stream) {
  const float* x  = (const float*)d_in[0];
  const float* Wq = (const float*)d_in[1];
  const float* bq = (const float*)d_in[2];
  const float* Wk = (const float*)d_in[3];
  const float* bk = (const float*)d_in[4];
  const float* Wv = (const float*)d_in[5];
  const float* bv = (const float*)d_in[6];
  float* out = (float*)d_out;

  // Workspace: Q (16 MB) | K (16 MB) | Vt (16 MB), all bf16
  unsigned short* Q  = (unsigned short*)d_ws;
  unsigned short* K  = Q + (size_t)ATT_M * ATT_D;
  unsigned short* Vt = K + (size_t)ATT_M * ATT_D;

  const dim3 blk(256);
  const dim3 gproj(ATT_M / 128, ATT_D / 64);          // 128 x 8
  qkv_proj_kernel<<<gproj, blk, 0, stream>>>(x, Wq, bq, Q, 0);
  qkv_proj_kernel<<<gproj, blk, 0, stream>>>(x, Wk, bk, K, 0);
  qkv_proj_kernel<<<gproj, blk, 0, stream>>>(x, Wv, bv, Vt, 1);

  const dim3 gattn(ATT_S / 32, 8);                    // 64 x 8
  flash_attn_kernel<<<gattn, blk, 0, stream>>>(Q, K, Vt, out);
}
